// DLS_13855564497183
// MI455X (gfx1250) — compile-verified, matrix operand order checked by reading, not run on hardware
//
#include <hip/hip_runtime.h>

typedef _Float16 v8h  __attribute__((ext_vector_type(8)));
typedef _Float16 v16h __attribute__((ext_vector_type(16)));
typedef float    v8f  __attribute__((ext_vector_type(8)));
typedef float    v4f  __attribute__((ext_vector_type(4)));

#define B_TOT 4096
#define SEQ   512
#define IN    8
#define HID   64
#define OUTD  4

#define WAVES 1
#define THREADS (WAVES * 32)
#define ROWS_PER_WAVE 16
#define ROWS_PER_BLOCK (WAVES * ROWS_PER_WAVE)   // 16
#define NBLOCKS (B_TOT / ROWS_PER_BLOCK)         // 256

// LDS layout (halves):
//   [0, 48*512)          : 48 weight B-fragments (kt=0..2, nt=0..15), 512 halves each
//   [48*512, +WAVES*1024): per-wave h scratch, 16x64 row-major f16
#define WFRAGS   48
#define W_HALVES (WFRAGS * 512)   // 24576
#define H_HALVES (WAVES * 1024)

#define LOG2E 1.4426950408889634f

// sigmoid(a + b) with bias pre-scaled: nb = -LOG2E*b, k = -LOG2E
__device__ __forceinline__ float sig_fused(float a, float k, float nb) {
    const float e = __builtin_amdgcn_exp2f(__builtin_fmaf(a, k, nb));
    return __builtin_amdgcn_rcpf(1.0f + e);
}

__global__ __launch_bounds__(THREADS, 1)
void lstm_fused_kernel(const float* __restrict__ x,
                       const float* __restrict__ W_ih,
                       const float* __restrict__ W_hh,
                       const float* __restrict__ b_ih,
                       const float* __restrict__ b_hh,
                       const float* __restrict__ W_fc,
                       const float* __restrict__ b_fc,
                       float* __restrict__ out)
{
    __shared__ __align__(128) _Float16 smem[W_HALVES + H_HALVES];

    const int tid  = threadIdx.x;
    const int lane = tid & 31;
    const int wave = tid >> 5;

    // ---------------------------------------------------------------
    // Pack weights [W_hh | W_ih | 0] (K=96, N=256) into B-fragment
    // layout in LDS. B frag (32x16, f16) per-lane layout mirrors the
    // 16-bit A layout: lane = n + 16*hi; VGPR v<4 holds Klocal =
    // hi*8 + 2v + {0,1}; v>=4 holds Klocal = 16 + hi*8 + 2(v-4)+{0,1}.
    // Per-lane data is 16 halves contiguous at frag*512 + lane*16.
    // ---------------------------------------------------------------
    for (int d = tid; d < WFRAGS * 256; d += THREADS) {
        const int frag = d >> 8;          // 0..47
        const int l    = (d >> 3) & 31;   // lane slot
        const int v    = d & 7;           // vgpr slot
        const int kt = frag >> 4, nt = frag & 15;
        const int n  = l & 15,   hi = l >> 4;
        #pragma unroll
        for (int h = 0; h < 2; ++h) {
            const int kk = (v < 4 ? 2 * v : 16 + 2 * (v - 4)) + hi * 8 + h;
            const int k  = kt * 32 + kk;
            const int g  = nt * 16 + n;
            float val = 0.0f;
            if (k < HID)                val = W_hh[g * HID + k];        // gates = h @ W_hh.T
            else if (k < HID + IN)      val = W_ih[g * IN + (k - HID)]; // + x @ W_ih.T
            smem[frag * 512 + l * 16 + v * 2 + h] = (_Float16)val;
        }
    }

    _Float16* hlds = smem + W_HALVES + wave * 1024;  // this wave's h: 16x64 row-major
    for (int i = lane; i < 1024; i += 32) hlds[i] = (_Float16)0.0f;

    __syncthreads();   // single-wave workgroup: trivial

    const int m0 = (blockIdx.x * WAVES + wave) * ROWS_PER_WAVE;
    const int n  = lane & 15;   // matrix row (A) / column (B,C,D) owned by this lane
    const int hi = lane >> 4;

    // Bias folded into the exponent-FMA constants of the gate nonlinearity:
    //   sigmoid gates (i,f,o):  exp2 arg = fma(acc, -LOG2E,   -LOG2E  *bias)
    //   tanh gate (g):          exp2 arg = fma(acc, -2*LOG2E, -2*LOG2E*bias)
    float nb1[16];
    #pragma unroll
    for (int nt = 0; nt < 16; ++nt) {
        const int g = nt * 16 + n;
        nb1[nt] = -LOG2E * (b_ih[g] + b_hh[g]);
    }
    float nb2[4];
    #pragma unroll
    for (int j = 0; j < 4; ++j) nb2[j] = 2.0f * nb1[8 + j];

    v8f c[4];
    #pragma unroll
    for (int j = 0; j < 4; ++j)
        #pragma unroll
        for (int r = 0; r < 8; ++r) c[j][r] = 0.0f;

    // A fragment kt=2: x occupies elements 0..7 of lanes 0..15 (K=64..71);
    // all other elements stay zero forever -> init once, rewrite only the
    // live part (predicated) inside the loop.
    v16h a2;
    #pragma unroll
    for (int i = 0; i < 16; ++i) a2[i] = (_Float16)0.0f;

    const float* xrow = x + (size_t)(m0 + n) * SEQ * IN;

    for (int s = 0; s < SEQ; ++s) {
        // ---- x tile -> live part of A fragment kt=2 ----
        if (hi == 0) {
            const float* xp = xrow + (size_t)s * IN;
            v4f x0 = *(const v4f*)xp;
            v4f x1 = *(const v4f*)(xp + 4);
            #pragma unroll
            for (int i = 0; i < 4; ++i) {
                a2[i]     = (_Float16)x0[i];
                a2[4 + i] = (_Float16)x1[i];
            }
        }

        // ---- A fragments kt=0,1: h from LDS (zeroed for s==0) ----
        v16h a01[2];
        #pragma unroll
        for (int kt = 0; kt < 2; ++kt) {
            const _Float16* hr = hlds + n * 64 + kt * 32 + hi * 8;
            v8h lo = *(const v8h*)(hr);
            v8h hh = *(const v8h*)(hr + 16);
            v16h a;
            #pragma unroll
            for (int i = 0; i < 8; ++i) { a[i] = lo[i]; a[8 + i] = hh[i]; }
            a01[kt] = a;
        }

        // ---- gates(16x256) = [h|x] @ [W_hh|W_ih]^T : 48 WMMAs, C0 = inline 0 ----
        v8f acc[16];
        #pragma unroll
        for (int nt = 0; nt < 16; ++nt) {
            v8f cz;
            #pragma unroll
            for (int i = 0; i < 8; ++i) cz[i] = 0.0f;
            v8f cc;
            v16h b0 = *(const v16h*)(smem + (0 * 16 + nt) * 512 + lane * 16);
            cc = __builtin_amdgcn_wmma_f32_16x16x32_f16(false, a01[0], false, b0,
                                                        (short)0, cz, false, false);
            v16h b1 = *(const v16h*)(smem + (1 * 16 + nt) * 512 + lane * 16);
            cc = __builtin_amdgcn_wmma_f32_16x16x32_f16(false, a01[1], false, b1,
                                                        (short)0, cc, false, false);
            v16h b2 = *(const v16h*)(smem + (2 * 16 + nt) * 512 + lane * 16);
            cc = __builtin_amdgcn_wmma_f32_16x16x32_f16(false, a2, false, b2,
                                                        (short)0, cc, false, false);
            acc[nt] = cc;
        }

        // ---- LSTM cell update (torch gate order i,f,g,o across nt groups) ----
        #pragma unroll
        for (int j = 0; j < 4; ++j) {
            #pragma unroll
            for (int r = 0; r < 8; ++r) {
                const float ig = sig_fused(acc[j][r],      -LOG2E,        nb1[j]);
                const float fg = sig_fused(acc[4 + j][r],  -LOG2E,        nb1[4 + j]);
                const float gg = 2.0f * sig_fused(acc[8 + j][r], -2.0f * LOG2E, nb2[j]) - 1.0f;
                const float og = sig_fused(acc[12 + j][r], -LOG2E,        nb1[12 + j]);
                const float cn = fg * c[j][r] + ig * gg;
                c[j][r] = cn;
                const float tc = 2.0f * sig_fused(cn, -2.0f * LOG2E, 0.0f) - 1.0f;
                const float hn = og * tc;
                const int row = r + hi * 8;                 // D layout: M = r + 8*hi
                hlds[row * 64 + j * 16 + n] = (_Float16)hn; // hidden k = j*16 + n
            }
        }
    }

    // ---- final FC (4x64) + softmax, one batch row per lane 0..15 ----
    if (hi == 0) {
        const _Float16* hr = hlds + n * 64;
        float o[OUTD];
        #pragma unroll
        for (int oo = 0; oo < OUTD; ++oo) {
            float a = b_fc[oo];
            #pragma unroll 8
            for (int k = 0; k < HID; ++k)
                a += (float)hr[k] * W_fc[oo * HID + k];
            o[oo] = a;
        }
        const float mx = fmaxf(fmaxf(o[0], o[1]), fmaxf(o[2], o[3]));
        float e[OUTD], ssum = 0.0f;
        #pragma unroll
        for (int oo = 0; oo < OUTD; ++oo) {
            e[oo] = __builtin_amdgcn_exp2f(LOG2E * (o[oo] - mx));
            ssum += e[oo];
        }
        const float rs = __builtin_amdgcn_rcpf(ssum);
        float* op = out + (size_t)(m0 + n) * OUTD;
        #pragma unroll
        for (int oo = 0; oo < OUTD; ++oo) op[oo] = e[oo] * rs;
    }
}

extern "C" void kernel_launch(void* const* d_in, const int* in_sizes, int n_in,
                              void* d_out, int out_size, void* d_ws, size_t ws_size,
                              hipStream_t stream) {
    (void)in_sizes; (void)n_in; (void)out_size; (void)d_ws; (void)ws_size;
    const float* x    = (const float*)d_in[0];
    const float* W_ih = (const float*)d_in[1];
    const float* W_hh = (const float*)d_in[2];
    const float* b_ih = (const float*)d_in[3];
    const float* b_hh = (const float*)d_in[4];
    const float* W_fc = (const float*)d_in[5];
    const float* b_fc = (const float*)d_in[6];
    float* out = (float*)d_out;

    lstm_fused_kernel<<<NBLOCKS, THREADS, 0, stream>>>(x, W_ih, W_hh, b_ih, b_hh,
                                                       W_fc, b_fc, out);
}
